// StandardAttention_5119601017042
// MI455X (gfx1250) — matrix operand order repaired and checked by reading, not compiled, not run on hardware
//
#include <hip/hip_runtime.h>

typedef __attribute__((ext_vector_type(16))) _Float16 v16h;
typedef __attribute__((ext_vector_type(8)))  _Float16 v8h;
typedef __attribute__((ext_vector_type(2)))  __fp16   fp16x2;
typedef __attribute__((ext_vector_type(8)))  float    v8f;
typedef __attribute__((ext_vector_type(4)))  float    v4f;
typedef __attribute__((ext_vector_type(8)))  int      v8i;
typedef __attribute__((ext_vector_type(4)))  int      v4i;

#define B_     4
#define H_     16
#define S_     2048
#define D_     64
#define QTILE  64                 // query rows per workgroup (16 per wave, 4 waves)
#define KTILE  64                 // keys per pipeline stage (2 softmax-amortized halves)
#define NTILES (S_ / KTILE)

static __device__ __forceinline__ v16h cat8(v8h lo, v8h hi) {
    return __builtin_shufflevector(lo, hi, 0,1,2,3,4,5,6,7,8,9,10,11,12,13,14,15);
}
// cross-half-wave (lane ^ 16) exchange via v_permlanex16_b32 (identity selects)
static __device__ __forceinline__ int xh_i(int v) {
    return __builtin_amdgcn_permlanex16(v, v, 0x76543210, 0xfedcba98, false, false);
}
static __device__ __forceinline__ float xh_f(float v) {
    int r = xh_i(__builtin_bit_cast(int, v));
    return __builtin_bit_cast(float, r);
}
static __device__ __forceinline__ int pack2h(float a, float b) {
    fp16x2 p = __builtin_amdgcn_cvt_pkrtz(a, b);
    return __builtin_bit_cast(int, p);
}

__global__ __launch_bounds__(128)
void StandardAttention_5119601017042_kernel(const float* __restrict__ Qg,
                                            const float* __restrict__ Kg,
                                            const float* __restrict__ Vg,
                                            float* __restrict__ Out)
{
    // Double-buffered LDS staging: K row-major (key,feat), V transposed (feat,key)
    __shared__ alignas(16) _Float16 Kt[2][KTILE][D_];    // 2 x 8 KB
    __shared__ alignas(16) _Float16 Vt[2][D_][KTILE];    // 2 x 8 KB

    const int tid   = threadIdx.x;
    const int wave  = tid >> 5;
    const int lane  = tid & 31;
    const int col   = lane & 15;     // query index (B/C col), or key/d row for A-frags
    const int hb    = lane >> 4;     // half-wave select
    const int hoff8 = hb * 8;        // C-layout row offset
    const int hb16  = hb * 16;       // B-frag K offset

    const int bh    = blockIdx.y;    // b*H + h
    const int b     = bh / H_;
    const int h     = bh % H_;
    const int qrow0 = blockIdx.x * QTILE + wave * 16;

    const float* Qp = Qg + (size_t)bh * S_ * D_;
    const float* Kp = Kg + (size_t)bh * S_ * D_;
    const float* Vp = Vg + (size_t)bh * S_ * D_;

    // ---- Q^T B-fragments: lane col = query, K-dim = features (contiguous in memory).
    //      scale = 1/sqrt(64) * log2(e) folded in (softmax runs in base-2 domain).
    const float qscale = 0.125f * 1.4426950408889634f;
    v16h qb[2];
    {
        const float* qr = Qp + (size_t)(qrow0 + col) * D_;
        #pragma unroll
        for (int c = 0; c < 2; ++c) {
            v8i words;
            #pragma unroll
            for (int j = 0; j < 4; ++j) {
                v4f f = *(const v4f*)(qr + c*32 + hb16 + 4*j);
                words[2*j]     = pack2h(f[0]*qscale, f[1]*qscale);
                words[2*j + 1] = pack2h(f[2]*qscale, f[3]*qscale);
            }
            qb[c] = __builtin_bit_cast(v16h, words);
        }
    }

    // ---- accumulators: O^T in C-layout (lane col = query, vgpr i -> d = t*16 + i + hoff8)
    v8f o[4];
    #pragma unroll
    for (int t = 0; t < 4; ++t)
        #pragma unroll
        for (int i = 0; i < 8; ++i) o[t][i] = 0.0f;
    float m_run = -__builtin_inff();
    float l_run = 0.0f;              // per-lane partial (own 32 keys); merged at end

    // ---- register staging for K/V tiles (software pipeline)
    //      each thread owns 2 adjacent keys x 16 features -> packed-pair LDS stores
    const int skey = (tid >> 2) << 1;    // even key 0..62
    const int sf   = (tid & 3) << 4;     // feature 0,16,32,48
    v4f k0[4], k1[4], vv0[4], vv1[4];
    {
        const float* kr = Kp + (size_t)skey * D_ + sf;
        const float* vr = Vp + (size_t)skey * D_ + sf;
        #pragma unroll
        for (int j = 0; j < 4; ++j) {
            k0[j]  = *(const v4f*)(kr + 4*j);        k1[j]  = *(const v4f*)(kr + D_ + 4*j);
            vv0[j] = *(const v4f*)(vr + 4*j);        vv1[j] = *(const v4f*)(vr + D_ + 4*j);
        }
    }

    for (int it = 0; it < NTILES; ++it) {
        _Float16 (*Kc)[D_]    = Kt[it & 1];
        _Float16 (*Vc)[KTILE] = Vt[it & 1];

        // stage current tile into LDS: K rows as packed b128; V transposed as packed b32
        {
            v4i ra, rb;
            ra[0] = pack2h(k0[0][0], k0[0][1]);  ra[1] = pack2h(k0[0][2], k0[0][3]);
            ra[2] = pack2h(k0[1][0], k0[1][1]);  ra[3] = pack2h(k0[1][2], k0[1][3]);
            rb[0] = pack2h(k0[2][0], k0[2][1]);  rb[1] = pack2h(k0[2][2], k0[2][3]);
            rb[2] = pack2h(k0[3][0], k0[3][1]);  rb[3] = pack2h(k0[3][2], k0[3][3]);
            *(v4i*)&Kc[skey][sf]     = ra;
            *(v4i*)&Kc[skey][sf + 8] = rb;
            ra[0] = pack2h(k1[0][0], k1[0][1]);  ra[1] = pack2h(k1[0][2], k1[0][3]);
            ra[2] = pack2h(k1[1][0], k1[1][1]);  ra[3] = pack2h(k1[1][2], k1[1][3]);
            rb[0] = pack2h(k1[2][0], k1[2][1]);  rb[1] = pack2h(k1[2][2], k1[2][3]);
            rb[2] = pack2h(k1[3][0], k1[3][1]);  rb[3] = pack2h(k1[3][2], k1[3][3]);
            *(v4i*)&Kc[skey + 1][sf]     = ra;
            *(v4i*)&Kc[skey + 1][sf + 8] = rb;
            #pragma unroll
            for (int j = 0; j < 4; ++j)
                #pragma unroll
                for (int e = 0; e < 4; ++e)
                    *(int*)&Vc[sf + 4*j + e][skey] = pack2h(vv0[j][e], vv1[j][e]); // keys (skey, skey+1)
        }
        // early-issue next tile's global loads (land during compute below)
        if (it + 1 < NTILES) {
            const int kb = (it + 1) * KTILE;
            const float* kr = Kp + (size_t)(kb + skey) * D_ + sf;
            const float* vr = Vp + (size_t)(kb + skey) * D_ + sf;
            if (kb + KTILE < S_) {           // global_prefetch_b8 two tiles ahead
                __builtin_prefetch(kr + (size_t)KTILE * D_, 0, 0);
                __builtin_prefetch(vr + (size_t)KTILE * D_, 0, 0);
            }
            #pragma unroll
            for (int j = 0; j < 4; ++j) {
                k0[j]  = *(const v4f*)(kr + 4*j);    k1[j]  = *(const v4f*)(kr + D_ + 4*j);
                vv0[j] = *(const v4f*)(vr + 4*j);    vv1[j] = *(const v4f*)(vr + D_ + 4*j);
            }
        }
        __syncthreads();   // single barrier/iter: buf[i&1] ready; also fences reads of buf from iter i-1

        // ---- S^T = K Q^T : four 16-key tiles; sT[t][i] = score(key = t*16 + i + hoff8, query = col)
        v8f sT[4];
        #pragma unroll
        for (int t = 0; t < 4; ++t) {
            v8f acc = {0.f,0.f,0.f,0.f,0.f,0.f,0.f,0.f};
            #pragma unroll
            for (int c = 0; c < 2; ++c) {
                const _Float16* krow = &Kc[t*16 + col][0];
                v8h lo = *(const v8h*)(krow + c*32 + hb16);
                v8h hi = *(const v8h*)(krow + c*32 + hb16 + 8);
                acc = __builtin_amdgcn_wmma_f32_16x16x32_f16(
                        false, cat8(lo, hi), false, qb[c], (short)0, acc, false, false);
            }
            sT[t] = acc;
        }

        // ---- online softmax: per-query stats are per-lane scalars (one permlanex16 to merge halves)
        float m8[8];
        #pragma unroll
        for (int i = 0; i < 8; ++i)
            m8[i] = fmaxf(fmaxf(sT[0][i], sT[1][i]), fmaxf(sT[2][i], sT[3][i]));
        float m4a = fmaxf(m8[0], m8[4]), m4b = fmaxf(m8[1], m8[5]);
        float m4c = fmaxf(m8[2], m8[6]), m4d = fmaxf(m8[3], m8[7]);
        float mloc = fmaxf(fmaxf(m4a, m4b), fmaxf(m4c, m4d));
        float mcur = fmaxf(mloc, xh_f(mloc));
        float mnew = fmaxf(m_run, mcur);
        // raw v_exp_f32: args are <= 0; underflow->0 is the desired softmax behavior
        float al   = __builtin_amdgcn_exp2f(m_run - mnew);
        m_run = mnew;

        #pragma unroll
        for (int t = 0; t < 4; ++t)
            #pragma unroll
            for (int i = 0; i < 8; ++i)
                sT[t][i] = __builtin_amdgcn_exp2f(sT[t][i] - mnew);

        float s8[8];
        #pragma unroll
        for (int i = 0; i < 8; ++i)
            s8[i] = (sT[0][i] + sT[1][i]) + (sT[2][i] + sT[3][i]);
        float s4a = s8[0] + s8[4], s4b = s8[1] + s8[5];
        float s4c = s8[2] + s8[6], s4d = s8[3] + s8[7];
        l_run = l_run * al + ((s4a + s4b) + (s4c + s4d));

        #pragma unroll
        for (int t = 0; t < 4; ++t)
            #pragma unroll
            for (int i = 0; i < 8; ++i) o[t][i] *= al;

        // ---- P^T B-fragments: pack to f16 pairs, swap words across half-waves.
        //      chunk c (keys 32c..32c+31) built from score tiles 2c, 2c+1
        int ow[4][4], xw[4][4];
        #pragma unroll
        for (int t = 0; t < 4; ++t)
            #pragma unroll
            for (int j = 0; j < 4; ++j)
                ow[t][j] = pack2h(sT[t][2*j], sT[t][2*j + 1]);
        #pragma unroll
        for (int t = 0; t < 4; ++t)
            #pragma unroll
            for (int j = 0; j < 4; ++j) xw[t][j] = xh_i(ow[t][j]);
        v16h pfrag[2];
        #pragma unroll
        for (int c = 0; c < 2; ++c) {
            v8i w;
            #pragma unroll
            for (int j = 0; j < 4; ++j) {
                w[j]     = hb ? xw[2*c + 1][j] : ow[2*c][j];     // keys 32c + hb16 + 0..7
                w[4 + j] = hb ? ow[2*c + 1][j] : xw[2*c][j];     // keys 32c + hb16 + 8..15
            }
            pfrag[c] = __builtin_bit_cast(v16h, w);
        }

        // ---- O^T += V^T P^T : four d-tiles, two k=32 WMMAs each
        #pragma unroll
        for (int t = 0; t < 4; ++t) {
            const _Float16* vrow = &Vc[t*16 + col][0];
            #pragma unroll
            for (int c = 0; c < 2; ++c) {
                v8h lo = *(const v8h*)(vrow + c*32 + hb16);
                v8h hi = *(const v8h*)(vrow + c*32 + hb16 + 8);
                o[t] = __builtin_amdgcn_wmma_f32_16x16x32_f16(
                           false, cat8(lo, hi), false, pfrag[c], (short)0, o[t], false, false);
            }
        }
    }

    // ---- finalize: merge denominator halves, scale, store d-contiguous (b128 stores)
    float linv = __builtin_amdgcn_rcpf(l_run + xh_f(l_run));
    float* op = Out + ((size_t)b * S_ + qrow0 + col) * (H_ * D_) + h * D_ + hoff8;
    #pragma unroll
    for (int t = 0; t < 4; ++t) {
        v4f r0, r1;
        #pragma unroll
        for (int e = 0; e < 4; ++e) { r0[e] = o[t][e] * linv; r1[e] = o[t][4 + e] * linv; }
        *(v4f*)(op + t*16)     = r0;
        *(v4f*)(op + t*16 + 4) = r1;
    }
}

extern "C" void kernel_launch(void* const* d_in, const int* in_sizes, int n_in,
                              void* d_out, int out_size, void* d_ws, size_t ws_size,
                              hipStream_t stream) {
    (void)in_sizes; (void)n_in; (void)out_size; (void)d_ws; (void)ws_size;
    const float* Q = (const float*)d_in[0];
    const float* K = (const float*)d_in[1];
    const float* V = (const float*)d_in[2];
    float* Out = (float*)d_out;
    dim3 grid(S_ / QTILE, B_ * H_);
    StandardAttention_5119601017042_kernel<<<grid, 128, 0, stream>>>(Q, K, V, Out);
}
